// FFT_84353157693551
// MI455X (gfx1250) — matrix operand order repaired and checked by reading, not compiled, not run on hardware
//
#include <hip/hip_runtime.h>
#include <hip/hip_bf16.h>
#include <math.h>

// ---------------------------------------------------------------------------
// Problem constants (match reference)
// ---------------------------------------------------------------------------
#define B_   2
#define C_   512
#define T_   2048
#define H_   8
#define DK_  64
#define FC_  2048
#define L_   2
#define EPS_ 1e-5f

typedef __attribute__((ext_vector_type(16))) _Float16 v16h;
typedef __attribute__((ext_vector_type(8)))  float    v8f;

// ---------------------------------------------------------------------------
// CDNA5 WMMA fragment index helpers (wave32, 16x16x32 f16 shapes)
// A matrix 16x32 (MxK), 16-bit: lane holds row m = lane&15.
//   half h -> v = h>>1, p = h&1 ; k = (v>=4?16:0) + 8*(lane>>4) + 2*(v&3) + p
// B matrix 32x16 (KxN), 16-bit: lane holds col n = lane&15.
//   half h -> k = (h>=8?16:0) + 8*(lane>>4) + (h&7)
// C/D 16x16 f32: vgpr r, lane -> m = r + 8*(lane>>4), n = lane&15
// ---------------------------------------------------------------------------
__device__ __forceinline__ int a_k_of(int h, int lane) {
  int v = h >> 1, p = h & 1;
  return ((v & 4) << 2) + ((lane >> 4) << 3) + ((v & 3) << 1) + p;
}
__device__ __forceinline__ int b_k_of(int h, int lane) {
  return ((h & 8) << 1) + ((lane >> 4) << 3) + (h & 7);
}

__device__ __forceinline__ float gelu_exact(float v) {
  return 0.5f * v * (1.0f + erff(v * 0.7071067811865475f));
}

// wave-local LDS fence (no block barrier: waves in attention diverge in trip
// count, so s_barrier would deadlock; DS ops are per-wave in-order but we
// still wait the counter and stop compiler reordering).
__device__ __forceinline__ void lds_fence() {
  asm volatile("s_wait_dscnt 0" ::: "memory");
}

// CDNA5 async copy: 16 bytes global -> LDS per lane, tracked by ASYNCcnt.
// GVS addressing: mem = saddr(u64) + vaddr(i32) + inst_offset.
__device__ __forceinline__ void async_copy_b128(unsigned lds_addr,
                                                unsigned gofs,
                                                unsigned long long gbase) {
  asm volatile("global_load_async_to_lds_b128 %0, %1, %2 offset:0"
               :: "v"(lds_addr), "v"(gofs), "s"(gbase) : "memory");
}
__device__ __forceinline__ void async_wait0() {
  asm volatile("s_wait_asynccnt 0" ::: "memory");
}

// ---------------------------------------------------------------------------
// GEMM:  Y[b,o,t] = act( scale * (sum_c W[o,c]*X[b,c,t] + bias[o]) ) * mask
// W: [O,K] row-major. X: [B,K,T_]. Y: [B,O,T_].
// grid = (T_/64, O/32, B), block = 256 (8 waves). Wave w owns the 16x16 tile
// at (M half = w&1, N group = w>>1) of the 32x64 block tile. K-step = 32.
// Tiles are DMA'd into LDS as fp32 via global_load_async_to_lds_b128 and
// converted to f16 at fragment-build time; next K tile is prefetched into L2
// while the async copies are in flight.
// act: 0 = none, 1 = exact GELU. mask: nullptr or [B,T_] broadcast multiply.
// ---------------------------------------------------------------------------
__global__ void __launch_bounds__(256)
gemm_bias_act(const float* __restrict__ W, const float* __restrict__ bias,
              const float* __restrict__ X, float* __restrict__ Y,
              const float* __restrict__ mask, int O, int K, int act, float scale)
{
  __shared__ float sWf[32][36];  // [m][k], rows 144 B (16B aligned)
  __shared__ float sXf[32][68];  // [k][n], rows 272 B (16B aligned)

  const int tid  = threadIdx.x;
  const int lane = tid & 31;
  const int wave = tid >> 5;
  const int mw   = wave & 1;        // 0..1  -> M offset 16*mw
  const int nw   = wave >> 1;       // 0..3  -> N offset 16*nw
  const int n0   = blockIdx.x * 64;
  const int o0   = blockIdx.y * 32;
  const int b    = blockIdx.z;
  const float* Xb = X + (size_t)b * K * T_;

  const unsigned long long wbase = (unsigned long long)(uintptr_t)W;
  const unsigned long long xbase = (unsigned long long)(uintptr_t)Xb;
  // generic shared addr = {SHARED_BASE, lds_offset}; low 32 bits = LDS offset
  const unsigned ldsW = (unsigned)(uintptr_t)&sWf[0][0];
  const unsigned ldsX = (unsigned)(uintptr_t)&sXf[0][0];

  // per-thread 16-byte transfer coordinates
  const int wr  = tid >> 3, wc = (tid & 7) << 2;    // W tile 32x32: 1 chunk
  const int xr0 = tid >> 4, xc = (tid & 15) << 2;   // X tile 32x64: 2 chunks

  v8f acc = {};

  for (int k0 = 0; k0 < K; k0 += 32) {
    // ---- async DMA of fp32 tiles straight into LDS (ASYNCcnt path) ----
    async_copy_b128(ldsW + (unsigned)(wr * 36 + wc) * 4u,
                    (unsigned)(((size_t)(o0 + wr) * K + (k0 + wc)) * 4u),
                    wbase);
    async_copy_b128(ldsX + (unsigned)(xr0 * 68 + xc) * 4u,
                    (unsigned)(((size_t)(k0 + xr0) * T_ + (n0 + xc)) * 4u),
                    xbase);
    async_copy_b128(ldsX + (unsigned)((xr0 + 16) * 68 + xc) * 4u,
                    (unsigned)(((size_t)(k0 + xr0 + 16) * T_ + (n0 + xc)) * 4u),
                    xbase);
    // ---- prefetch next K tile into L2 while the DMA runs ----
    if (k0 + 32 < K) {
      __builtin_prefetch(&W[(size_t)(o0 + wr) * K + (k0 + 32 + wc)], 0, 1);
      __builtin_prefetch(&Xb[(size_t)(k0 + 32 + xr0) * T_ + (n0 + xc)], 0, 1);
    }
    async_wait0();
    __syncthreads();

    v16h a, bf;
#pragma unroll
    for (int h = 0; h < 16; ++h) {
      a[h]  = (_Float16)sWf[16 * mw + (lane & 15)][a_k_of(h, lane)];
      bf[h] = (_Float16)sXf[b_k_of(h, lane)][16 * nw + (lane & 15)];
    }
    acc = __builtin_amdgcn_wmma_f32_16x16x32_f16(false, a, false, bf,
                                                 (short)0, acc, false, false);
    __syncthreads();
  }

#pragma unroll
  for (int r = 0; r < 8; ++r) {
    int row = o0 + 16 * mw + r + ((lane >> 4) << 3);
    int col = n0 + 16 * nw + (lane & 15);
    float v = (acc[r] + bias[row]) * scale;
    if (act == 1) v = gelu_exact(v);
    if (mask) v *= mask[b * T_ + col];
    Y[((size_t)b * O + row) * T_ + col] = v;
  }
}

// ---------------------------------------------------------------------------
// Causal multi-head attention with proximal bias, flash-style online softmax.
// Q/K/V/O all stored [B, C_, T_] (channel c = h*DK + d). Q is pre-scaled.
// block = 256 (8 waves); each wave independently handles one 16-query tile of
// one (b,h). Key blocks of 32; per block: 4 QK^T WMMAs + 4 PV WMMAs.
// Softmax bookkeeping is wave-local in LDS (lanes 0..15 own the 16 rows).
// ---------------------------------------------------------------------------
__global__ void __launch_bounds__(256)
attn_causal(const float* __restrict__ Q, const float* __restrict__ Kb,
            const float* __restrict__ V, float* __restrict__ Ob)
{
  __shared__ float    sS[8][16][33];   // scores (fp32, bias+mask applied)
  __shared__ _Float16 sP[8][16][34];   // probabilities (f16, A-operand source)
  __shared__ float    sM[8][16];       // running row max
  __shared__ float    sL[8][16];       // running row sum
  __shared__ float    sC[8][16];       // per-block rescale factor

  const int tid  = threadIdx.x;
  const int lane = tid & 31;
  const int w    = tid >> 5;
  const int tile = blockIdx.x * 8 + w;
  const int ntq  = T_ / 16;
  const int bh   = tile / ntq;
  const int q0   = (tile % ntq) * 16;
  const int b    = bh / H_;
  const int h2   = bh % H_;
  const size_t base = ((size_t)b * C_ + h2 * DK_) * T_;  // start of this head

  // Q fragments (A layout), two 32-wide d-steps covering D=64
  v16h qa0, qa1;
  {
    int m = lane & 15;
#pragma unroll
    for (int h = 0; h < 16; ++h) {
      int kk = a_k_of(h, lane);
      qa0[h] = (_Float16)Q[base + (size_t)kk * T_ + q0 + m];
      qa1[h] = (_Float16)Q[base + (size_t)(32 + kk) * T_ + q0 + m];
    }
  }

  if (lane < 16) { sM[w][lane] = -__builtin_inff(); sL[w][lane] = 0.f; }
  lds_fence();

  v8f oaccs[4] = {};  // 16x64 output tile, four 16-wide d groups

  for (int s0 = 0; s0 <= q0 + 15; s0 += 32) {
    // prefetch the next key block's K rows into L2
    if (s0 + 32 <= q0 + 15) {
      __builtin_prefetch(&Kb[base + (size_t)(lane & 15) * T_ + s0 + 32], 0, 1);
      __builtin_prefetch(&V[base + (size_t)(lane & 15) * T_ + s0 + 32], 0, 1);
    }
    // ---- S = Q K^T over this 32-key block ----
    v8f sc0 = {}, sc1 = {};
#pragma unroll
    for (int ds = 0; ds < 2; ++ds) {
      v16h kb0, kb1;
#pragma unroll
      for (int h = 0; h < 16; ++h) {
        int kk = b_k_of(h, lane);                      // d within this d-step
        size_t rowb = base + (size_t)(ds * 32 + kk) * T_;
        kb0[h] = (_Float16)Kb[rowb + s0 + (lane & 15)];
        kb1[h] = (_Float16)Kb[rowb + s0 + 16 + (lane & 15)];
      }
      v16h qa = ds ? qa1 : qa0;
      sc0 = __builtin_amdgcn_wmma_f32_16x16x32_f16(false, qa, false, kb0,
                                                   (short)0, sc0, false, false);
      sc1 = __builtin_amdgcn_wmma_f32_16x16x32_f16(false, qa, false, kb1,
                                                   (short)0, sc1, false, false);
    }

    // ---- proximal bias + causal mask -> LDS (fp32) ----
#pragma unroll
    for (int r = 0; r < 8; ++r) {
      int m  = r + ((lane >> 4) << 3);
      int gi = q0 + m;
      int n  = lane & 15;
      int j0 = s0 + n;
      float v0 = sc0[r] - fabsf((float)(gi - j0));
      if (j0 > gi) v0 = -__builtin_inff();
      sS[w][m][n] = v0;
      int j1 = s0 + 16 + n;
      float v1 = sc1[r] - fabsf((float)(gi - j1));
      if (j1 > gi) v1 = -__builtin_inff();
      sS[w][m][16 + n] = v1;
    }
    lds_fence();

    // ---- online softmax update: lanes 0..15 each own one row ----
    if (lane < 16) {
      int m = lane;
      float mo = sM[w][m];
      float mx = mo;
#pragma unroll
      for (int j = 0; j < 32; ++j) mx = fmaxf(mx, sS[w][m][j]);
      float csc = expf(mo - mx);   // exp(-inf) = 0 on first block
      float sum = 0.f;
#pragma unroll
      for (int j = 0; j < 32; ++j) {
        float p = expf(sS[w][m][j] - mx);
        sP[w][m][j] = (_Float16)p;
        sum += p;
      }
      sL[w][m] = sL[w][m] * csc + sum;
      sM[w][m] = mx;
      sC[w][m] = csc;
    }
    lds_fence();

    // ---- rescale running output accumulators ----
#pragma unroll
    for (int r = 0; r < 8; ++r) {
      float f = sC[w][r + ((lane >> 4) << 3)];
      oaccs[0][r] *= f; oaccs[1][r] *= f; oaccs[2][r] *= f; oaccs[3][r] *= f;
    }

    // ---- O += P V ----
    v16h pa;
#pragma unroll
    for (int h = 0; h < 16; ++h) pa[h] = sP[w][lane & 15][a_k_of(h, lane)];
#pragma unroll
    for (int g = 0; g < 4; ++g) {
      v16h vb;
#pragma unroll
      for (int h = 0; h < 16; ++h) {
        int kk = b_k_of(h, lane);                      // key within block
        vb[h] = (_Float16)V[base + (size_t)(g * 16 + (lane & 15)) * T_ + s0 + kk];
      }
      oaccs[g] = __builtin_amdgcn_wmma_f32_16x16x32_f16(false, pa, false, vb,
                                                        (short)0, oaccs[g],
                                                        false, false);
    }
  }

  lds_fence();
#pragma unroll
  for (int r = 0; r < 8; ++r) {
    int m = r + ((lane >> 4) << 3);
    float inv = 1.0f / sL[w][m];
    int t = q0 + m;
#pragma unroll
    for (int g = 0; g < 4; ++g) {
      int d = g * 16 + (lane & 15);
      Ob[base + (size_t)d * T_ + t] = oaccs[g][r] * inv;
    }
  }
}

// ---------------------------------------------------------------------------
// Out[b,:,t] = LN_c(X[b,:,t] + Ybuf[b,:,t]) * g + bb, optionally * mask[b,t]
// grid = B*T_, block = 256 (C_ = 512 -> 2 channels/thread)
// ---------------------------------------------------------------------------
__global__ void __launch_bounds__(256)
add_layernorm(const float* __restrict__ X, const float* __restrict__ Ybuf,
              const float* __restrict__ g, const float* __restrict__ bb,
              const float* __restrict__ mask, float* __restrict__ Out)
{
  __shared__ float s1[256], s2[256];
  const int bt = blockIdx.x;
  const int b  = bt / T_, t = bt % T_;
  const size_t base = (size_t)b * C_ * T_ + t;
  const int c0 = threadIdx.x, c1 = threadIdx.x + 256;

  float v0 = X[base + (size_t)c0 * T_] + Ybuf[base + (size_t)c0 * T_];
  float v1 = X[base + (size_t)c1 * T_] + Ybuf[base + (size_t)c1 * T_];
  s1[threadIdx.x] = v0 + v1;
  s2[threadIdx.x] = v0 * v0 + v1 * v1;
  __syncthreads();
  for (int off = 128; off > 0; off >>= 1) {
    if (threadIdx.x < off) {
      s1[threadIdx.x] += s1[threadIdx.x + off];
      s2[threadIdx.x] += s2[threadIdx.x + off];
    }
    __syncthreads();
  }
  float mean = s1[0] * (1.0f / C_);
  float var  = s2[0] * (1.0f / C_) - mean * mean;   // biased, matches jnp.var
  float rstd = rsqrtf(var + EPS_);
  float mk   = mask ? mask[bt] : 1.0f;
  Out[base + (size_t)c0 * T_] = ((v0 - mean) * rstd * g[c0] + bb[c0]) * mk;
  Out[base + (size_t)c1 * T_] = ((v1 - mean) * rstd * g[c1] + bb[c1]) * mk;
}

// ---------------------------------------------------------------------------
// xb = x * x_mask  (mask is [B,1,T])
// ---------------------------------------------------------------------------
__global__ void __launch_bounds__(256)
mask_mul(const float* __restrict__ x, const float* __restrict__ mask,
         float* __restrict__ out)
{
  size_t i = (size_t)blockIdx.x * 256 + threadIdx.x;
  if (i >= (size_t)B_ * C_ * T_) return;
  int t = (int)(i % T_);
  int b = (int)(i / ((size_t)C_ * T_));
  out[i] = x[i] * mask[b * T_ + t];
}

// ---------------------------------------------------------------------------
// Host orchestration
// ---------------------------------------------------------------------------
extern "C" void kernel_launch(void* const* d_in, const int* in_sizes, int n_in,
                              void* d_out, int out_size, void* d_ws, size_t ws_size,
                              hipStream_t stream) {
  (void)in_sizes; (void)n_in; (void)out_size; (void)ws_size;

  const float* x    = (const float*)d_in[0];
  const float* xm   = (const float*)d_in[1];
  const float* wq   = (const float*)d_in[2];
  const float* bq   = (const float*)d_in[3];
  const float* wk   = (const float*)d_in[4];
  const float* bk   = (const float*)d_in[5];
  const float* wv   = (const float*)d_in[6];
  const float* bv   = (const float*)d_in[7];
  const float* wo   = (const float*)d_in[8];
  const float* bo   = (const float*)d_in[9];
  const float* ln0g = (const float*)d_in[10];
  const float* ln0b = (const float*)d_in[11];
  const float* w1   = (const float*)d_in[12];
  const float* b1   = (const float*)d_in[13];
  const float* w2   = (const float*)d_in[14];
  const float* b2   = (const float*)d_in[15];
  const float* ln1g = (const float*)d_in[16];
  const float* ln1b = (const float*)d_in[17];
  float* out = (float*)d_out;

  const size_t BCT = (size_t)B_ * C_ * T_;
  float* ws = (float*)d_ws;
  float* xb = ws; ws += BCT;   // running activations
  float* x1 = ws; ws += BCT;   // post-attention LN
  float* qb = ws; ws += BCT;
  float* kb = ws; ws += BCT;
  float* vb = ws; ws += BCT;
  float* ob = ws; ws += BCT;   // attention output
  float* yb = ws; ws += BCT;   // projection / ffn2 output
  float* hb = ws;              // [B, FC_, T_] ffn hidden

  dim3 blk(256);
  mask_mul<<<dim3((unsigned)((BCT + 255) / 256)), blk, 0, stream>>>(x, xm, xb);

  const dim3 gproj(T_ / 64, C_ / 32, B_);
  const dim3 gffn1(T_ / 64, FC_ / 32, B_);
  const dim3 gattn(B_ * H_ * (T_ / 16) / 8);
  const dim3 gln(B_ * T_);
  const float qscale = 0.125f;  // DK^-0.5 = 64^-0.5

  for (int i = 0; i < L_; ++i) {
    const float* wqi = wq + (size_t)i * C_ * C_;
    const float* wki = wk + (size_t)i * C_ * C_;
    const float* wvi = wv + (size_t)i * C_ * C_;
    const float* woi = wo + (size_t)i * C_ * C_;
    const float* w1i = w1 + (size_t)i * FC_ * C_;
    const float* w2i = w2 + (size_t)i * C_ * FC_;

    gemm_bias_act<<<gproj, blk, 0, stream>>>(wqi, bq + i * C_, xb, qb, nullptr,
                                             C_, C_, 0, qscale);
    gemm_bias_act<<<gproj, blk, 0, stream>>>(wki, bk + i * C_, xb, kb, nullptr,
                                             C_, C_, 0, 1.0f);
    gemm_bias_act<<<gproj, blk, 0, stream>>>(wvi, bv + i * C_, xb, vb, nullptr,
                                             C_, C_, 0, 1.0f);
    attn_causal<<<gattn, blk, 0, stream>>>(qb, kb, vb, ob);
    gemm_bias_act<<<gproj, blk, 0, stream>>>(woi, bo + i * C_, ob, yb, nullptr,
                                             C_, C_, 0, 1.0f);
    add_layernorm<<<gln, blk, 0, stream>>>(xb, yb, ln0g + i * C_, ln0b + i * C_,
                                           nullptr, x1);
    gemm_bias_act<<<gffn1, blk, 0, stream>>>(w1i, b1 + i * FC_, x1, hb, nullptr,
                                             FC_, C_, 1, 1.0f);
    gemm_bias_act<<<gproj, blk, 0, stream>>>(w2i, b2 + i * C_, hb, yb, xm,
                                             C_, FC_, 0, 1.0f);
    const bool last = (i == L_ - 1);
    add_layernorm<<<gln, blk, 0, stream>>>(x1, yb, ln1g + i * C_, ln1b + i * C_,
                                           last ? xm : nullptr,
                                           last ? out : xb);
  }
}
